// FastEdgeNetwork_19645180412350
// MI455X (gfx1250) — matrix operand order repaired and checked by reading, not compiled, not run on hardware
//
#include <hip/hip_runtime.h>

// MI455X / gfx1250: wave32, WMMA bf16 16x16x32, f32 accumulate.
typedef __attribute__((ext_vector_type(16))) __bf16 v16bf;
typedef __attribute__((ext_vector_type(8)))  float  v8f;

union FragB {
  v16bf v;
  unsigned u[8];
  uint4 q[2];
};

__device__ __forceinline__ unsigned short f2bf(float f) {
  // round-to-nearest-even fp32 -> bf16
  unsigned u = __float_as_uint(f);
  unsigned r = (u + 0x7FFFu + ((u >> 16) & 1u)) >> 16;
  return (unsigned short)r;
}

// ---------------------------------------------------------------------------
// Pack conv weights [32][32][3][3] fp32 into per-lane WMMA B-fragment layout:
// wpack[t][ntile][lane][j] (u32 = bf16 pair), t = kh*3+kw, K-step t covers
// ci 0..31.  B 32x16 bf16 in this code's K order: lane holds column n=lane&15; lanes 0-15
// carry K 0..15, lanes 16-31 carry K 16..31; VGPR j holds K = kbase+2j,+1.
// ---------------------------------------------------------------------------
__global__ void pack_weights_kernel(const float* __restrict__ w_node,
                                    const float* __restrict__ w_edge,
                                    unsigned* __restrict__ wpack) {
  int idx = blockIdx.x * 256 + threadIdx.x;     // 2 * 4608 total
  if (idx >= 2 * 4608) return;
  int which = idx / 4608;
  int rem   = idx - which * 4608;
  const float* w = which ? w_edge : w_node;
  int j     = rem & 7;
  int lane  = (rem >> 3) & 31;
  int ntile = (rem >> 8) & 1;
  int t     = rem >> 9;                         // 0..8
  int co = ntile * 16 + (lane & 15);
  int ci = ((lane & 16) ? 16 : 0) + 2 * j;
  // w layout [co][ci][kh][kw]: flat = (co*32+ci)*9 + t
  unsigned lo = f2bf(w[(co * 32 + ci    ) * 9 + t]);
  unsigned hi = f2bf(w[(co * 32 + ci + 1) * 9 + t]);
  wpack[idx] = lo | (hi << 16);
}

// ---------------------------------------------------------------------------
// One workgroup (8 waves) = one item (node or edge).
// Implicit-GEMM 3x3 conv: M=64 pixels, N=32 cout, K=288 (9 taps x 32 ci).
// EDGE=false: out = conv(x, w)          (writes th table)
// EDGE=true : out = elu(th[src] * conv(x, w))
// ---------------------------------------------------------------------------
template <bool EDGE>
__global__ __launch_bounds__(256)
void conv_wmma_kernel(const float* __restrict__ x,        // [N][32][8][8]
                      const unsigned* __restrict__ wpack, // 4608 u32 B-frags
                      const float* __restrict__ thTab,    // [4096][2048] (EDGE)
                      const int* __restrict__ srcIdx,     // [E] (EDGE)
                      float* __restrict__ out) {          // [N][2048]
  // Padded bf16 input tile, channel-last: cell(ph,pw) holds 32 bf16 = 16 u32,
  // padded to 17 u32/cell so frag reads are bank-conflict free.
  __shared__ unsigned ldsTile[100 * 17];
  __shared__ float    ldsOut[32 * 65];          // [co][pix], pad 65 for banks

  const int item = blockIdx.x;
  const int tid  = threadIdx.x;
  const int lane = tid & 31;
  const int wv   = tid >> 5;

  const float* xRow = x + (size_t)item * 2048;
  const float* thRow = nullptr;
  if (EDGE) {
    thRow = thTab + (size_t)srcIdx[item] * 2048;
    __builtin_prefetch(thRow, 0, 3);            // global_prefetch_b8 (L2-hot)
  }

  // --- Phase 1: zero pad ring, then stage bf16 tile in LDS ---
  for (int i = tid; i < 100 * 17; i += 256) ldsTile[i] = 0u;
  __syncthreads();
  {
    unsigned short* lds16 = (unsigned short*)ldsTile;
    for (int i = tid; i < 2048; i += 256) {     // coalesced global read
      int ci = i >> 6, p = i & 63, h = p >> 3, w = p & 7;
      lds16[((h + 1) * 10 + (w + 1)) * 34 + ci] = f2bf(xRow[i]);
    }
  }
  __syncthreads();

  // --- Phase 2: 9 K-steps of v_wmma_f32_16x16x32_bf16 per wave ---
  const int mtile = wv >> 1;                    // 0..3 (16 pixels each)
  const int ntile = wv & 1;                     // 0..1 (16 cout each)
  const int m  = mtile * 16 + (lane & 15);      // pixel index
  const int hh = m >> 3, ww = m & 7;
  const int hi4 = (lane & 16) ? 8 : 0;          // A frag: lanes 16-31 -> K+16

  v8f acc = {};
#pragma unroll
  for (int t = 0; t < 9; ++t) {
    const int kh = t / 3, kw = t - 3 * (t / 3);
    const int abase = ((hh + kh) * 10 + (ww + kw)) * 17 + hi4;
    FragB a, b;
#pragma unroll
    for (int j = 0; j < 4; ++j) {               // A 16x32 bf16 frag layout
      a.u[j]     = ldsTile[abase + j];          // K = base+2j,+1
      a.u[4 + j] = ldsTile[abase + 4 + j];      // K = base+8+2j,+1
    }
    const uint4* bq =
        (const uint4*)wpack + ((size_t)(t * 2 + ntile) * 32 + lane) * 2;
    b.q[0] = bq[0];
    b.q[1] = bq[1];
    acc = __builtin_amdgcn_wmma_f32_16x16x32_bf16(
        /*neg_a=*/false, a.v, /*neg_b=*/false, b.v,
        /*c_mod=*/(short)0, acc, /*reuse_a=*/false, /*reuse_b=*/false);
  }

  // --- Phase 3: transpose C frags through LDS, fused epilogue, coalesced out
  {
    // C/D layout: lane 0-15: N=lane, M=r; lane 16-31: N=lane-16, M=r+8
    const int n    = ntile * 16 + (lane & 15);
    const int mrow = mtile * 16 + ((lane & 16) ? 8 : 0);
#pragma unroll
    for (int r = 0; r < 8; ++r) ldsOut[n * 65 + mrow + r] = acc[r];
  }
  __syncthreads();

  float* outRow = out + (size_t)item * 2048;
  for (int i = tid; i < 2048; i += 256) {
    int co = i >> 6, p = i & 63;
    float z = ldsOut[co * 65 + p];
    if (EDGE) {
      float v = thRow[i] * z;                   // L2-resident gather
      outRow[i] = v > 0.f ? v : expm1f(v);      // elu, alpha = 1
    } else {
      outRow[i] = z;
    }
  }
}

// ---------------------------------------------------------------------------
extern "C" void kernel_launch(void* const* d_in, const int* in_sizes, int n_in,
                              void* d_out, int out_size, void* d_ws,
                              size_t ws_size, hipStream_t stream) {
  const float* inp  = (const float*)d_in[0];  // [4096][32][8][8]
  const int*   esrc = (const int*)  d_in[1];  // [32768]
  const float* e    = (const float*)d_in[2];  // [32768][32][8][8]
  const float* wn   = (const float*)d_in[3];  // [32][32][3][3]
  const float* we   = (const float*)d_in[4];  // [32][32][3][3]
  float* out = (float*)d_out;

  const int n_nodes = in_sizes[0] / 2048;
  const int n_edges = in_sizes[1];

  // ws: [0, 4608) wpack_node | [4608, 9216) wpack_edge | th table (33.5 MB)
  unsigned* wpack = (unsigned*)d_ws;
  float* thTab = (float*)(wpack + 2 * 4608);

  pack_weights_kernel<<<(2 * 4608 + 255) / 256, 256, 0, stream>>>(wn, we,
                                                                  wpack);
  conv_wmma_kernel<false><<<n_nodes, 256, 0, stream>>>(
      inp, wpack, nullptr, nullptr, thTab);
  conv_wmma_kernel<true><<<n_edges, 256, 0, stream>>>(
      e, wpack + 4608, thTab, esrc, out);
}
